// SOMFNN_25907242730005
// MI455X (gfx1250) — matrix operand order repaired and checked by reading, not compiled, not run on hardware
//
#include <hip/hip_runtime.h>
#include <hip/hip_bf16.h>

// ---------------------------------------------------------------------------
// SOMFNN fused kernel for gfx1250 (MI455X):
//   out[s,o] = sum_r lambda[s,r] * sigmoid( X[s,:] . W[r*512+o,:] + b[r*512+o] )
// S=8192, D_IN=1024, R=16, O=512.  Compute-bound (AI ~1700 FLOP/B) -> bf16 WMMA,
// double-buffered LDS fed by gfx1250 async global->LDS DMA (ASYNCcnt),
// bank-conflict-free padded LDS rows, 2-deep B-fragment register pipeline.
// ---------------------------------------------------------------------------

typedef __attribute__((ext_vector_type(16))) __bf16 v16bf;
typedef __attribute__((ext_vector_type(4)))  __bf16 v4bf;
typedef __attribute__((ext_vector_type(8)))  float  v8f;
typedef __attribute__((ext_vector_type(4)))  int    v4i;

#define S_DIM   8192
#define D_IN    1024
#define N_RULES 16
#define N_OUT   512

#define BM 64          // s rows per block   (4 waves x 16)
#define BN 32          // o cols per block   (2 waves x 16)
#define BK 32          // k per WMMA step (16x16x32 bf16)
#define KSTEPS (D_IN / BK)      // 32
#define LSTR   40      // padded LDS row stride in bf16 (20 DWORDs, 16B aligned,
                       // {20n mod 64} distinct for n=0..15 -> no 4-way conflicts)

// --- gfx1250 async global->LDS DMA (tracked by ASYNCcnt) ------------------
__device__ __forceinline__ void async_ld_b128(unsigned lds_off, const void* gaddr) {
  asm volatile("global_load_async_to_lds_b128 %0, %1, off"
               :: "v"(lds_off), "v"(gaddr) : "memory");
}
__device__ __forceinline__ void wait_async0() {
  asm volatile("s_wait_asynccnt 0x0" ::: "memory");
}
// Flat pointer low 32 bits == wave-relative LDS byte offset on AMDGPU.
__device__ __forceinline__ unsigned lds_off(const void* p) {
  return (unsigned)(unsigned long long)p;
}

// --- fp32 -> bf16 conversion (RNE), 4 elements / thread -------------------
__global__ __launch_bounds__(256) void cvt_f32_bf16(const float* __restrict__ src,
                                                    __bf16* __restrict__ dst,
                                                    int n4) {
  int i = blockIdx.x * blockDim.x + threadIdx.x;
  if (i >= n4) return;
  float4 f = ((const float4*)src)[i];
  v4bf o;
  o[0] = (__bf16)f.x;
  o[1] = (__bf16)f.y;
  o[2] = (__bf16)f.z;
  o[3] = (__bf16)f.w;
  ((v4bf*)dst)[i] = o;
}

// --- fused GEMM + sigmoid + rule-strength reduction -----------------------
__global__ __launch_bounds__(256, 1) void somfnn_fused(
    const __bf16* __restrict__ Xb,   // [S, D_IN]       bf16
    const __bf16* __restrict__ Wb,   // [R*O, D_IN]     bf16
    const float*  __restrict__ bias, // [R*O]
    const float*  __restrict__ lam,  // [S, R]
    float*        __restrict__ out)  // [S, O]
{
  // Double-buffered LDS with padded rows: W chunk for ALL 16 rules + X chunk.
  __shared__ __bf16 WsBuf[2][N_RULES * BN * LSTR];  // 2 x 40 KB : [r*32+oo][kk(pad)]
  __shared__ __bf16 XsBuf[2][BM * LSTR];            // 2 x  5 KB : [row][kk(pad)]

  const int t    = threadIdx.x;
  const int lane = t & 31;        // wave32 on gfx1250
  const int wave = t >> 5;        // 8 waves
  const int wvs  = wave & 3;      // s sub-tile 0..3
  const int wvo  = wave >> 2;     // o sub-tile 0..1

  const int o0 = blockIdx.x * BN;
  const int s0 = blockIdx.y * BM;

  const int n  = lane & 15;       // C/D layout: N = lane&15
  const int hi = lane >> 4;       // C/D layout: M = 8*hi + j

  const int ocol = o0 + wvo * 16 + n;

  // ---- per-thread DMA geometry (constant across K loop) ------------------
  // X: 64 rows x 4 kparts = 256 slots (1 per thread), 16B each.
  const int xrow = t >> 2, xkp = t & 3;
  const __bf16* xsrc = Xb + (size_t)(s0 + xrow) * D_IN + xkp * 8;
  const unsigned xdst = lds_off(&XsBuf[0][xrow * LSTR + xkp * 8]);
  // W: 16 rules x 32 rows x 4 kparts = 2048 slots (8 per thread), 16B each.
  const __bf16* wsrc[8];
  unsigned      wdst[8];
#pragma unroll
  for (int i = 0; i < 8; ++i) {
    int slot = i * 256 + t;
    int rowi = slot >> 2;          // r*32 + oo
    int kp   = slot & 3;
    int r    = rowi >> 5;
    int oo   = rowi & 31;
    wsrc[i] = Wb + (size_t)(r * N_OUT + o0 + oo) * D_IN + kp * 8;
    wdst[i] = lds_off(&WsBuf[0][rowi * LSTR + kp * 8]);
  }
  const unsigned xstride = (unsigned)sizeof(XsBuf[0]);
  const unsigned wstride = (unsigned)sizeof(WsBuf[0]);

  // ---- init accumulators with bias (bias depends on column only) ---------
  v8f acc[N_RULES];
#pragma unroll
  for (int r = 0; r < N_RULES; ++r) {
    float bv = bias[r * N_OUT + ocol];
#pragma unroll
    for (int j = 0; j < 8; ++j) acc[r][j] = bv;
  }

  // Per-wave B-fragment base inside a buffer (elements).
  const int bbase = (wvo * 16 + n) * LSTR + hi * 16;
  const int abase = (wvs * 16 + n) * LSTR + hi * 16;

  // ---- prologue: kick DMA for chunk 0 into buffer 0 ----------------------
  {
    async_ld_b128(xdst, xsrc);
#pragma unroll
    for (int i = 0; i < 8; ++i) async_ld_b128(wdst[i], wsrc[i]);
  }

  // ---- main K loop: compute chunk `step` while chunk `step+1` streams in -
#pragma unroll 2
  for (int step = 0; step < KSTEPS; ++step) {
    const int buf = step & 1;

    wait_async0();        // my DMA for this buffer landed (in-order)
    __syncthreads();      // everyone's DMA landed; prev compute done

    if (step + 1 < KSTEPS) {      // refill the buffer consumed last step
      const int kc = (step + 1) * BK;
      const unsigned bo = (unsigned)(buf ^ 1);
      async_ld_b128(xdst + bo * xstride, xsrc + kc);
#pragma unroll
      for (int i = 0; i < 8; ++i)
        async_ld_b128(wdst[i] + bo * wstride, wsrc[i] + kc);
    }

    // --- A fragment (16x32 bf16, contiguous k order this code uses for both A and B):
    //  lanes 0-15 : M=lane,    K 0..15
    //  lanes16-31 : M=lane-16, K 16..31
    v16bf afrag;
    {
      const __bf16* ap = &XsBuf[buf][abase];
      ((v4i*)&afrag)[0] = *(const v4i*)(ap);
      ((v4i*)&afrag)[1] = *(const v4i*)(ap + 8);
    }

    // --- B fragments + WMMA for all 16 rules, 2-deep register pipeline.
    //  B (32x16 bf16, same contiguous k order as A, not the ISA table order): lanes 0-15 N=lane, K 0..15 ; lanes 16-31 N=lane-16, K 16..31
    const __bf16* wsb = &WsBuf[buf][bbase];
    v16bf b0, b1;
    ((v4i*)&b0)[0] = *(const v4i*)(wsb);
    ((v4i*)&b0)[1] = *(const v4i*)(wsb + 8);
#pragma unroll
    for (int r = 0; r < N_RULES; ++r) {
      if (r + 1 < N_RULES) {      // prefetch fragment r+1 into the other slot
        const __bf16* bp = wsb + (r + 1) * (BN * LSTR);
        v16bf& dstf = (r & 1) ? b0 : b1;
        ((v4i*)&dstf)[0] = *(const v4i*)(bp);
        ((v4i*)&dstf)[1] = *(const v4i*)(bp + 8);
      }
      const v16bf& use = (r & 1) ? b1 : b0;
      acc[r] = __builtin_amdgcn_wmma_f32_16x16x32_bf16(
          /*neg_a=*/false, afrag, /*neg_b=*/false, use,
          /*c_mod=*/(short)0, acc[r], /*reuse_a=*/false, /*reuse_b=*/false);
    }
  }

  // ---- epilogue: sigmoid + lambda-weighted reduction over rules ----------
#pragma unroll
  for (int j = 0; j < 8; ++j) {
    const int srow = s0 + wvs * 16 + hi * 8 + j;   // C/D layout: M = 8*hi + j
    const float* lp = lam + (size_t)srow * N_RULES;
    float o = 0.f;
#pragma unroll
    for (int r = 0; r < N_RULES; ++r) {
      float h = __builtin_amdgcn_rcpf(1.0f + __expf(-acc[r][j]));  // v_rcp_f32
      o += lp[r] * h;
    }
    out[(size_t)srow * N_OUT + ocol] = o;
  }
}

// ---------------------------------------------------------------------------
extern "C" void kernel_launch(void* const* d_in, const int* in_sizes, int n_in,
                              void* d_out, int out_size, void* d_ws, size_t ws_size,
                              hipStream_t stream) {
  const float* X   = (const float*)d_in[0];   // [8192,1024]
  const float* W   = (const float*)d_in[1];   // [8192,1024]
  const float* b   = (const float*)d_in[2];   // [8192]
  const float* lam = (const float*)d_in[3];   // [8192,16]
  float* out = (float*)d_out;                 // [8192,512]

  __bf16* Xb = (__bf16*)d_ws;                       // 16 MB
  __bf16* Wb = Xb + (size_t)S_DIM * D_IN;           // 16 MB

  const int n4 = (S_DIM * D_IN) / 4;                // X and W have equal size
  cvt_f32_bf16<<<(n4 + 255) / 256, 256, 0, stream>>>(X, Xb, n4);
  cvt_f32_bf16<<<(n4 + 255) / 256, 256, 0, stream>>>(W, Wb, n4);

  dim3 grid(N_OUT / BN, S_DIM / BM);  // 16 x 128
  somfnn_fused<<<grid, 256, 0, stream>>>(Xb, Wb, b, lam, out);
}
